// VRNN_7224134992464
// MI455X (gfx1250) — matrix operand order repaired and checked by reading, not compiled, not run on hardware
//
#include <hip/hip_runtime.h>
#include <hip/hip_bf16.h>
#include <math.h>

// ---------------------------------------------------------------------------
// VRNN loss on gfx1250 (CDNA5, wave32, WMMA f32 <- f16xf16 16x16x32)
// B=128, T=96, D=800, L=128, S=10
// f16 operands staged in workspace (weights pre-converted once per launch);
// f32 accumulate. Bulk GEMMs use 32x32-per-wave register blocking
// (2x2 fragments -> 1 vector load per WMMA).
// ---------------------------------------------------------------------------

typedef __attribute__((ext_vector_type(16))) _Float16 v16h;
typedef __attribute__((ext_vector_type(8)))  _Float16 v8h;
typedef __attribute__((ext_vector_type(8)))  float    v8f;

#define B_SZ 128
#define T_SZ 96
#define D_SZ 800
#define L_SZ 128
#define S_SZ 10

__device__ __forceinline__ float softplus_f(float x) {
  return (x > 20.f) ? x : log1pf(expf(x));
}
__device__ __forceinline__ float sigmoid_f(float x) {
  return 1.f / (1.f + expf(-x));
}

__device__ __forceinline__ v8f wmma16(v16h a, v16h b, v8f c) {
  // D = A(16x32 f16) * B(32x16 f16) + C(16x16 f32)
  return __builtin_amdgcn_wmma_f32_16x16x32_f16(false, a, false, b, (short)0, c,
                                                false, false);
}

// A fragment (16x32, MxK, row-major f16 source).
// CDNA5 layout: lane m=ln&15; lanes 0-15 hold K {0..7,16..23}, lanes 16-31
// hold K {8..15,24..31}; VGPR v holds halves 2v,2v+1. Two 16B vector loads.
__device__ __forceinline__ v16h frag_a_h(const _Float16* A, size_t lda, int m0, int k0) {
  int ln = threadIdx.x & 31;
  const _Float16* p = A + (size_t)(m0 + (ln & 15)) * lda + (size_t)k0 + ((ln >> 4) << 3);
  v8h lo = *(const v8h*)p;
  v8h hi = *(const v8h*)(p + 16);
  return __builtin_shufflevector(lo, hi, 0, 1, 2, 3, 4, 5, 6, 7,
                                 8, 9, 10, 11, 12, 13, 14, 15);
}

// B fragment (32x16, KxN) from row-major f16 W[N][K] (computes X @ W^T).
// Lane n=ln&15 reads W row n: lanes 0-15 K {0..7,16..23}, lanes 16-31 K {8..15,24..31}.
// Two 16B vector loads.
__device__ __forceinline__ v16h frag_b_h(const _Float16* W, size_t ldw, int n0, int k0) {
  int ln = threadIdx.x & 31;
  const _Float16* p = W + (size_t)(n0 + (ln & 15)) * ldw + (size_t)k0 + ((ln >> 4) << 3);
  v8h lo = *(const v8h*)p;
  v8h hi = *(const v8h*)(p + 16);
  return __builtin_shufflevector(lo, hi, 0, 1, 2, 3, 4, 5, 6, 7,
                                 8, 9, 10, 11, 12, 13, 14, 15);
}

// C/D 16x16 f32 layout: element i -> row i + 8*(lane>=16), col lane&15.
__device__ __forceinline__ int out_row(int i) {
  return i + (((threadIdx.x & 31) >> 4) << 3);
}
__device__ __forceinline__ int out_col() { return threadIdx.x & 15; }

// 32x32-per-wave GEMM core: 2x2 fragment blocking, 1 vector load per WMMA.
__device__ __forceinline__ void gemm32(const _Float16* __restrict__ A, size_t lda,
                                       const _Float16* __restrict__ W, size_t ldw,
                                       int tm, int tn, int kdim, v8f acc[2][2]) {
  for (int k0 = 0; k0 < kdim; k0 += 32) {
    v16h a0 = frag_a_h(A, lda, tm * 32, k0);
    v16h a1 = frag_a_h(A, lda, tm * 32 + 16, k0);
    v16h b0 = frag_b_h(W, ldw, tn * 32, k0);
    v16h b1 = frag_b_h(W, ldw, tn * 32 + 16, k0);
    acc[0][0] = wmma16(a0, b0, acc[0][0]);
    acc[0][1] = wmma16(a0, b1, acc[0][1]);
    acc[1][0] = wmma16(a1, b0, acc[1][0]);
    acc[1][1] = wmma16(a1, b1, acc[1][1]);
  }
}

// ---------------------------------------------------------------------------
__global__ void k_f32to16(const float* __restrict__ src, _Float16* __restrict__ dst,
                          int n) {
  int i = blockIdx.x * blockDim.x + threadIdx.x;
  if (i < n) dst[i] = (_Float16)src[i];
}

// xt_h[b][d] = (f16) inputs[b][t][d]
__global__ void k_xt(const float* __restrict__ inputs, int t,
                     _Float16* __restrict__ xt) {
  int i = blockIdx.x * blockDim.x + threadIdx.x;
  if (i < B_SZ * D_SZ) {
    int b = i / D_SZ, d = i - b * D_SZ;
    xt[i] = (_Float16)inputs[(size_t)b * T_SZ * D_SZ + (size_t)t * D_SZ + d];
  }
}

__global__ void k_init(float* __restrict__ c_s, float* __restrict__ elbo,
                       _Float16* __restrict__ out_h, _Float16* __restrict__ h_h) {
  int i = blockIdx.x * blockDim.x + threadIdx.x;
  if (i < B_SZ * L_SZ) {
    c_s[i] = 0.f;
    out_h[i] = (_Float16)0.f;
    h_h[i] = (_Float16)0.f;
  }
  if (i < B_SZ) elbo[i] = 0.f;
}

// x_hat = relu(xt @ W_px^T + b_px); also builds enc_in = [out, x_hat] (f16).
// M=128 N=128 K=800 -> 4x4 32-tiles -> grid 4 (4 waves/block).
__global__ void k_phix(const _Float16* __restrict__ xt,
                       const _Float16* __restrict__ W_px, const float* __restrict__ b_px,
                       const _Float16* __restrict__ out_h,
                       _Float16* __restrict__ x_hat, _Float16* __restrict__ enc_in) {
  int tile = blockIdx.x * 4 + (threadIdx.x >> 5);
  int tm = tile >> 2, tn = tile & 3;
  v8f acc[2][2] = {};
  gemm32(xt, D_SZ, W_px, D_SZ, tm, tn, D_SZ, acc);
#pragma unroll
  for (int tj = 0; tj < 2; ++tj) {
    int n = tn * 32 + tj * 16 + out_col();
    float bias = b_px[n];
#pragma unroll
    for (int ti = 0; ti < 2; ++ti)
#pragma unroll
      for (int i = 0; i < 8; ++i) {
        int m = tm * 32 + ti * 16 + out_row(i);
        _Float16 v = (_Float16)fmaxf(acc[ti][tj][i] + bias, 0.f);
        x_hat[m * L_SZ + n] = v;
        enc_in[m * 2 * L_SZ + L_SZ + n] = v;
        enc_in[m * 2 * L_SZ + n] = out_h[m * L_SZ + n];  // left half = out
      }
  }
}

// ph = relu(out @ W_pr^T + b_pr); mu_p = ph[:, :L];
// sigma_p = softplus(exp(ph[:, L:]) + 0.5). M=128 N=256 K=128 -> 4x8 -> grid 8.
__global__ void k_prior(const _Float16* __restrict__ out_h,
                        const _Float16* __restrict__ W_pr, const float* __restrict__ b_pr,
                        float* __restrict__ mu_p, float* __restrict__ sigma_p) {
  int tile = blockIdx.x * 4 + (threadIdx.x >> 5);
  int tm = tile >> 3, tn = tile & 7;
  v8f acc[2][2] = {};
  gemm32(out_h, L_SZ, W_pr, L_SZ, tm, tn, L_SZ, acc);
#pragma unroll
  for (int tj = 0; tj < 2; ++tj) {
    int n = tn * 32 + tj * 16 + out_col();
    float bias = b_pr[n];
#pragma unroll
    for (int ti = 0; ti < 2; ++ti)
#pragma unroll
      for (int i = 0; i < 8; ++i) {
        int m = tm * 32 + ti * 16 + out_row(i);
        float v = fmaxf(acc[ti][tj][i] + bias, 0.f);
        if (n < L_SZ) mu_p[m * L_SZ + n] = v;
        else          sigma_p[m * L_SZ + n - L_SZ] = softplus_f(expf(v) + 0.5f);
      }
  }
}

// eh = relu(enc_in @ W_en^T + b_en); mu_q = eh[:, :L] + mu_p;
// sigma_q = exp(eh[:, L:]). M=128 N=256 K=256 -> 4x8 -> grid 8.
__global__ void k_encoder(const _Float16* __restrict__ enc_in,
                          const _Float16* __restrict__ W_en, const float* __restrict__ b_en,
                          const float* __restrict__ mu_p,
                          float* __restrict__ mu_q, float* __restrict__ sigma_q) {
  int tile = blockIdx.x * 4 + (threadIdx.x >> 5);
  int tm = tile >> 3, tn = tile & 7;
  v8f acc[2][2] = {};
  gemm32(enc_in, 2 * L_SZ, W_en, 2 * L_SZ, tm, tn, 2 * L_SZ, acc);
#pragma unroll
  for (int tj = 0; tj < 2; ++tj) {
    int n = tn * 32 + tj * 16 + out_col();
    float bias = b_en[n];
#pragma unroll
    for (int ti = 0; ti < 2; ++ti)
#pragma unroll
      for (int i = 0; i < 8; ++i) {
        int m = tm * 32 + ti * 16 + out_row(i);
        float v = fmaxf(acc[ti][tj][i] + bias, 0.f);
        if (n < L_SZ) mu_q[m * L_SZ + n] = v + mu_p[m * L_SZ + n];
        else          sigma_q[m * L_SZ + n - L_SZ] = expf(v);
      }
  }
}

// z = mu_q + sigma_q*eps (stored f16 for phi_z); kl[b,s] = sum_l(log_qz-log_pz)
// in f32; fills dec_in[:, L:2L] = out, rnn_in[:, :L] = x_hat; zeroes log_px.
// grid = B*S blocks, 128 threads (l).
__global__ void k_sample(const float* __restrict__ eps_t,
                         const float* __restrict__ mu_q, const float* __restrict__ sigma_q,
                         const float* __restrict__ mu_p, const float* __restrict__ sigma_p,
                         const _Float16* __restrict__ out_h,
                         const _Float16* __restrict__ x_hat,
                         _Float16* __restrict__ z_h, _Float16* __restrict__ dec_in,
                         _Float16* __restrict__ rnn_in,
                         float* __restrict__ kl, float* __restrict__ log_px) {
  int bs = blockIdx.x;
  int b  = bs / S_SZ;
  int l  = threadIdx.x;
  float mq = mu_q[b * L_SZ + l], sq = sigma_q[b * L_SZ + l];
  float mp = mu_p[b * L_SZ + l], sp = sigma_p[b * L_SZ + l];
  float e  = eps_t[(size_t)bs * L_SZ + l];
  float zv = mq + sq * e;
  z_h[bs * L_SZ + l] = (_Float16)zv;
  dec_in[bs * 2 * L_SZ + L_SZ + l] = out_h[b * L_SZ + l];
  rnn_in[bs * 2 * L_SZ + l] = x_hat[b * L_SZ + l];
  float dp = (zv - mp) / sp;
  // (-0.5*eps^2 - ln sq) - (-0.5*dp^2 - ln sp); 0.5*LOG2PI cancels
  float v = (-0.5f * e * e - logf(sq)) - (-0.5f * dp * dp - logf(sp));
  __shared__ float red[L_SZ];
  red[l] = v;
  __syncthreads();
  for (int s = 64; s > 0; s >>= 1) {
    if (l < s) red[l] += red[l + s];
    __syncthreads();
  }
  if (l == 0) { kl[bs] = red[0]; log_px[bs] = 0.f; }
}

// z_hat = relu(z @ W_pz^T + b_pz); writes dec_in[:, :L], rnn_in[:, L:2L].
// M=1280 N=128 K=128 -> 40x4 32-tiles -> grid 40.
__global__ void k_phiz(const _Float16* __restrict__ z_h,
                       const _Float16* __restrict__ W_pz, const float* __restrict__ b_pz,
                       _Float16* __restrict__ dec_in, _Float16* __restrict__ rnn_in) {
  int tile = blockIdx.x * 4 + (threadIdx.x >> 5);
  int tm = tile >> 2, tn = tile & 3;
  v8f acc[2][2] = {};
  gemm32(z_h, L_SZ, W_pz, L_SZ, tm, tn, L_SZ, acc);
#pragma unroll
  for (int tj = 0; tj < 2; ++tj) {
    int n = tn * 32 + tj * 16 + out_col();
    float bias = b_pz[n];
#pragma unroll
    for (int ti = 0; ti < 2; ++ti)
#pragma unroll
      for (int i = 0; i < 8; ++i) {
        int m = tm * 32 + ti * 16 + out_row(i);
        _Float16 v = (_Float16)fmaxf(acc[ti][tj][i] + bias, 0.f);
        dec_in[m * 2 * L_SZ + n] = v;
        rnn_in[m * 2 * L_SZ + L_SZ + n] = v;
      }
  }
}

// logits = relu(dec_in @ W_de^T + b_de); log_px[m] += sum_n(l*x - softplus(l)).
// M=1280 N=800 K=256 -> 40x25 32-tiles -> grid 250. Row-sum shfl_xor + atomics.
__global__ void k_decoder(const _Float16* __restrict__ dec_in,
                          const _Float16* __restrict__ W_de, const float* __restrict__ b_de,
                          const float* __restrict__ inputs, int t,
                          float* __restrict__ log_px) {
  int tile = blockIdx.x * 4 + (threadIdx.x >> 5);
  int tm = tile / 25, tn = tile % 25;
  // warm weight panel into cache (global_prefetch_b8 path)
  __builtin_prefetch((const void*)(W_de + (size_t)(tn * 32) * 2 * L_SZ), 0, 1);
  v8f acc[2][2] = {};
  gemm32(dec_in, 2 * L_SZ, W_de, 2 * L_SZ, tm, tn, 2 * L_SZ, acc);
#pragma unroll
  for (int tj = 0; tj < 2; ++tj) {
    int n = tn * 32 + tj * 16 + out_col();
    float bias = b_de[n];
#pragma unroll
    for (int ti = 0; ti < 2; ++ti)
#pragma unroll
      for (int i = 0; i < 8; ++i) {
        int m = tm * 32 + ti * 16 + out_row(i);
        int b = m / S_SZ;
        float logit = fmaxf(acc[ti][tj][i] + bias, 0.f);
        float xv = inputs[(size_t)b * T_SZ * D_SZ + (size_t)t * D_SZ + n];
        float term = logit * xv - softplus_f(logit);
#pragma unroll
        for (int w = 1; w < 16; w <<= 1) term += __shfl_xor(term, w, 32);
        if ((threadIdx.x & 15) == 0) atomicAdd(&log_px[m], term);
      }
  }
}

// Fused per-step LSTM over S=10 samples + elbo accumulation + out_new.
// Batch partitioned: block handles 16 rows; no cross-block dependency.
// Gates GEMM: M=16, N=512, K=256 (rnn_in) + 128 (h f16 in LDS). 8 waves x 4 tiles,
// A fragments hoisted (reused across the 4 N-tiles).
__global__ void k_lstm(const _Float16* __restrict__ rnn_in,
                       const _Float16* __restrict__ W_ih, const _Float16* __restrict__ W_hh,
                       const float* __restrict__ b_ih, const float* __restrict__ b_hh,
                       const float* __restrict__ log_px, const float* __restrict__ kl,
                       const float* __restrict__ beta,
                       _Float16* __restrict__ h_g, float* __restrict__ c_g,
                       _Float16* __restrict__ out_h, float* __restrict__ elbo_sum) {
  __shared__ __align__(32) _Float16 sh_h[16 * L_SZ];  // 4 KB
  __shared__ float sh_g[16 * 4 * L_SZ];               // 32 KB
  int b0 = blockIdx.x * 16;
  int tid = threadIdx.x;
  int wave = tid >> 5;

  if (blockIdx.x == 0 && tid < B_SZ) {  // elbo_sum[b] += mean_s(log_px - beta*kl)
    float a = 0.f, bt = beta[0];
#pragma unroll
    for (int s = 0; s < S_SZ; ++s)
      a += log_px[tid * S_SZ + s] - bt * kl[tid * S_SZ + s];
    elbo_sum[tid] += a * (1.f / S_SZ);
  }

  float creg[8], oacc[8];
  int ebase = tid * 8;  // 256 threads * 8 = 16*128 elements
#pragma unroll
  for (int j = 0; j < 8; ++j) {
    int e = ebase + j, r = e >> 7, l = e & 127;
    sh_h[e] = h_g[(b0 + r) * L_SZ + l];
    creg[j] = c_g[(b0 + r) * L_SZ + l];
    oacc[j] = 0.f;
  }
  __syncthreads();

  for (int s = 0; s < S_SZ; ++s) {
    // relative row r maps to rnn_in[(b0+r)*S + s], ld = S*2L
    const _Float16* A1 = rnn_in + (size_t)(b0 * S_SZ + s) * 2 * L_SZ;
    v16h a1[8], a2[4];
#pragma unroll
    for (int kc = 0; kc < 8; ++kc)
      a1[kc] = frag_a_h(A1, (size_t)S_SZ * 2 * L_SZ, 0, kc * 32);
#pragma unroll
    for (int kc = 0; kc < 4; ++kc)
      a2[kc] = frag_a_h(sh_h, L_SZ, 0, kc * 32);

#pragma unroll
    for (int jt = 0; jt < 4; ++jt) {
      int tn = wave * 4 + jt;
      v8f acc = {};
#pragma unroll
      for (int kc = 0; kc < 8; ++kc)
        acc = wmma16(a1[kc], frag_b_h(W_ih, 2 * L_SZ, tn * 16, kc * 32), acc);
#pragma unroll
      for (int kc = 0; kc < 4; ++kc)
        acc = wmma16(a2[kc], frag_b_h(W_hh, L_SZ, tn * 16, kc * 32), acc);
      int n = tn * 16 + out_col();
      float bias = b_ih[n] + b_hh[n];
#pragma unroll
      for (int i = 0; i < 8; ++i)
        sh_g[out_row(i) * 4 * L_SZ + n] = acc[i] + bias;
    }
    __syncthreads();

#pragma unroll
    for (int j = 0; j < 8; ++j) {
      int e = ebase + j, r = e >> 7, l = e & 127;
      float gi = sh_g[r * 4 * L_SZ + l];
      float gf = sh_g[r * 4 * L_SZ + L_SZ + l];
      float gg = sh_g[r * 4 * L_SZ + 2 * L_SZ + l];
      float go = sh_g[r * 4 * L_SZ + 3 * L_SZ + l];
      float cv = sigmoid_f(gf) * creg[j] + sigmoid_f(gi) * tanhf(gg);
      creg[j] = cv;
      float hv = sigmoid_f(go) * tanhf(cv);
      oacc[j] += hv;
      sh_h[e] = (_Float16)hv;
    }
    __syncthreads();
  }

#pragma unroll
  for (int j = 0; j < 8; ++j) {
    int e = ebase + j, r = e >> 7, l = e & 127;
    h_g[(b0 + r) * L_SZ + l] = sh_h[e];
    c_g[(b0 + r) * L_SZ + l] = creg[j];
    out_h[(b0 + r) * L_SZ + l] = (_Float16)(oacc[j] * (1.f / S_SZ));
  }
}

__global__ void k_final(const float* __restrict__ elbo_sum, float* __restrict__ outp) {
  __shared__ float red[B_SZ];
  int i = threadIdx.x;
  red[i] = elbo_sum[i] * (1.f / T_SZ);
  __syncthreads();
  for (int s = 64; s > 0; s >>= 1) {
    if (i < s) red[i] += red[i + s];
    __syncthreads();
  }
  if (i == 0) outp[0] = -red[0] * (1.f / B_SZ);
}

// ---------------------------------------------------------------------------
extern "C" void kernel_launch(void* const* d_in, const int* in_sizes, int n_in,
                              void* d_out, int out_size, void* d_ws, size_t ws_size,
                              hipStream_t stream) {
  const float* inputs = (const float*)d_in[0];
  const float* beta   = (const float*)d_in[1];
  const float* eps    = (const float*)d_in[2];
  const float* W_px = (const float*)d_in[3];  const float* b_px = (const float*)d_in[4];
  const float* W_pz = (const float*)d_in[5];  const float* b_pz = (const float*)d_in[6];
  const float* W_pr = (const float*)d_in[7];  const float* b_pr = (const float*)d_in[8];
  const float* W_en = (const float*)d_in[9];  const float* b_en = (const float*)d_in[10];
  const float* W_de = (const float*)d_in[11]; const float* b_de = (const float*)d_in[12];
  const float* W_ih = (const float*)d_in[13]; const float* W_hh = (const float*)d_in[14];
  const float* b_ih = (const float*)d_in[15]; const float* b_hh = (const float*)d_in[16];

  const int BL = B_SZ * L_SZ;  // 16384

  // ---- f32 region ----
  float* ws = (float*)d_ws;
  float* mu_p    = ws;              // [B,L]
  float* sigma_p = mu_p + BL;       // [B,L]
  float* mu_q    = sigma_p + BL;    // [B,L]
  float* sigma_q = mu_q + BL;       // [B,L]
  float* c_s     = sigma_q + BL;    // [B,L]
  float* kl      = c_s + BL;        // [B*S]
  float* log_px  = kl + B_SZ * S_SZ;     // [B*S]
  float* elbo    = log_px + B_SZ * S_SZ; // [B]
  float* f32_end = elbo + B_SZ;

  // ---- f16 region (32B-aligned; all sizes multiples of 16 elements) ----
  _Float16* hb      = (_Float16*)f32_end;
  _Float16* out_h   = hb;                 // [B,L]
  _Float16* h_h     = out_h + BL;         // [B,L]
  _Float16* x_hat   = h_h + BL;           // [B,L]
  _Float16* enc_in  = x_hat + BL;         // [B,2L]
  _Float16* z_h     = enc_in + 2 * BL;    // [B*S,L]
  _Float16* dec_in  = z_h + S_SZ * BL;    // [B*S,2L]
  _Float16* rnn_in  = dec_in + 2 * S_SZ * BL;  // [B*S,2L]
  _Float16* xt_h    = rnn_in + 2 * S_SZ * BL;  // [B,D]
  _Float16* Wpx_h   = xt_h + B_SZ * D_SZ;      // [L,D]
  _Float16* Wpz_h   = Wpx_h + L_SZ * D_SZ;     // [L,L]
  _Float16* Wpr_h   = Wpz_h + L_SZ * L_SZ;     // [2L,L]
  _Float16* Wen_h   = Wpr_h + 2 * L_SZ * L_SZ; // [2L,2L]
  _Float16* Wde_h   = Wen_h + 4 * L_SZ * L_SZ; // [D,2L]
  _Float16* Wih_h   = Wde_h + D_SZ * 2 * L_SZ; // [4L,2L]
  _Float16* Whh_h   = Wih_h + 8 * L_SZ * L_SZ; // [4L,L]

  // one-time (per launch) weight conversion f32 -> f16
  auto cvt = [&](const float* s, _Float16* d, int n) {
    k_f32to16<<<(n + 255) / 256, 256, 0, stream>>>(s, d, n);
  };
  cvt(W_px, Wpx_h, L_SZ * D_SZ);
  cvt(W_pz, Wpz_h, L_SZ * L_SZ);
  cvt(W_pr, Wpr_h, 2 * L_SZ * L_SZ);
  cvt(W_en, Wen_h, 4 * L_SZ * L_SZ);
  cvt(W_de, Wde_h, D_SZ * 2 * L_SZ);
  cvt(W_ih, Wih_h, 8 * L_SZ * L_SZ);
  cvt(W_hh, Whh_h, 4 * L_SZ * L_SZ);

  k_init<<<(BL + 255) / 256, 256, 0, stream>>>(c_s, elbo, out_h, h_h);

  for (int t = 0; t < T_SZ; ++t) {
    const float* eps_t = eps + (size_t)t * B_SZ * S_SZ * L_SZ;
    k_xt<<<(B_SZ * D_SZ + 255) / 256, 256, 0, stream>>>(inputs, t, xt_h);
    k_phix<<<4, 128, 0, stream>>>(xt_h, Wpx_h, b_px, out_h, x_hat, enc_in);
    k_prior<<<8, 128, 0, stream>>>(out_h, Wpr_h, b_pr, mu_p, sigma_p);
    k_encoder<<<8, 128, 0, stream>>>(enc_in, Wen_h, b_en, mu_p, mu_q, sigma_q);
    k_sample<<<B_SZ * S_SZ, 128, 0, stream>>>(eps_t, mu_q, sigma_q, mu_p, sigma_p,
                                              out_h, x_hat, z_h, dec_in, rnn_in,
                                              kl, log_px);
    k_phiz<<<40, 128, 0, stream>>>(z_h, Wpz_h, b_pz, dec_in, rnn_in);
    k_decoder<<<250, 128, 0, stream>>>(dec_in, Wde_h, b_de, inputs, t, log_px);
    k_lstm<<<8, 256, 0, stream>>>(rnn_in, Wih_h, Whh_h, b_ih, b_hh, log_px, kl, beta,
                                  h_h, c_s, out_h, elbo);
  }

  k_final<<<1, 128, 0, stream>>>(elbo, (float*)d_out);
}
